// AE1_45397804319349
// MI455X (gfx1250) — compile-verified, matrix operand order checked by reading, not run on hardware
//
#include <hip/hip_runtime.h>
#include <hip/hip_bf16.h>

// ---------------------------------------------------------------------------
// Stacked LSTM encoder/decoder for MI455X (gfx1250), wave32 + WMMA bf16.
//   encoder: 5 layers, in 1280 -> hid 256
//   decoder: 5 layers, in 256  -> hid 1280
// One fused kernel per (layer, timestep): input GEMM + recurrent GEMM +
// bias + gate activations + cell update.
//   block = 128 threads = 4 waves; wave g owns gate g (i,f,g,o)
//   each wave computes TWO 16x16 WMMA tiles (32 hidden columns); fragments
//   are software-pipelined at distance 2 (triple-buffered) so every WMMA
//   consumes fragments loaded a full iteration earlier.
// All GEMM operands bf16 (pre-cast once), f32 accumulation.
// ---------------------------------------------------------------------------

#define DEV __device__ __forceinline__

typedef __attribute__((ext_vector_type(16))) __bf16 bf16x16;
typedef __attribute__((ext_vector_type(8)))  float  v8f;

union FragU {
  bf16x16 v;
  uint4   q[2];
};

DEV unsigned short f32_to_bf16(float f) {
  unsigned int u = __float_as_uint(f);
  u += 0x7FFFu + ((u >> 16) & 1u);      // round-to-nearest-even
  return (unsigned short)(u >> 16);
}

// Load one 16x32 bf16 fragment row for this lane. A- and B-fragments use the
// same lane -> (row, k-subset) mapping for 16-bit data:
//   half = lane>>4 :  k in {k0+8h .. k0+8h+7} U {k0+16+8h .. k0+16+8h+7}
DEV bf16x16 load_frag(const unsigned short* __restrict__ rowPtr, int k0, int half) {
  FragU fu;
  const unsigned short* p = rowPtr + k0 + 8 * half;
  fu.q[0] = *(const uint4*)(p);
  fu.q[1] = *(const uint4*)(p + 16);
  return fu.v;
}

DEV v8f wmma_bf16(bf16x16 a, bf16x16 b, v8f c) {
  return __builtin_amdgcn_wmma_f32_16x16x32_bf16(
      false, a, false, b, (short)0, c, false, false);
}

// Distance-2 software-pipelined dual-tile GEMM accumulation over K:
//   acc0 += A @ B0-tile,  acc1 += A @ B1-tile
// Fragments for chunks k and k+32 stay resident while chunk k+64 is fetched,
// so no WMMA ever has to drain the load queue (loadcnt stays >= 6).
DEV void gemm2_pipelined(v8f& acc0, v8f& acc1,
                         const unsigned short* __restrict__ aRow,
                         const unsigned short* __restrict__ b0Row,
                         const unsigned short* __restrict__ b1Row,
                         int K, int half) {
  bf16x16 a0  = load_frag(aRow,  0, half);
  bf16x16 b00 = load_frag(b0Row, 0, half);
  bf16x16 b10 = load_frag(b1Row, 0, half);
  if (K <= 32) {
    acc0 = wmma_bf16(a0, b00, acc0);
    acc1 = wmma_bf16(a0, b10, acc1);
    return;
  }
  bf16x16 a1  = load_frag(aRow,  32, half);
  bf16x16 b01 = load_frag(b0Row, 32, half);
  bf16x16 b11 = load_frag(b1Row, 32, half);
  for (int k0 = 0; k0 + 64 < K; k0 += 32) {
    bf16x16 an  = load_frag(aRow,  k0 + 64, half);
    bf16x16 b0n = load_frag(b0Row, k0 + 64, half);
    bf16x16 b1n = load_frag(b1Row, k0 + 64, half);
    acc0 = wmma_bf16(a0, b00, acc0);
    acc1 = wmma_bf16(a0, b10, acc1);
    a0 = a1;  b00 = b01; b10 = b11;
    a1 = an;  b01 = b0n; b11 = b1n;
  }
  acc0 = wmma_bf16(a0, b00, acc0);
  acc1 = wmma_bf16(a0, b10, acc1);
  acc0 = wmma_bf16(a1, b01, acc0);
  acc1 = wmma_bf16(a1, b11, acc1);
}

// ---------------------------------------------------------------------------
// Fused LSTM timestep.
//   grid = (hid/32, B/16), block = 128
//   gates = in_t @ W_ih^T + h_prev @ W_hh^T + bias   (PyTorch order i,f,g,o)
//   c' = sig(f)*c + sig(i)*tanh(g);  h' = sig(o)*tanh(c')
// ---------------------------------------------------------------------------
__global__ __launch_bounds__(128) void lstm_step_kernel(
    const unsigned short* __restrict__ in_t,     // [B, in_dim] bf16 slice @ t
    const unsigned short* __restrict__ hprev,    // [B, hid] bf16
    const unsigned short* __restrict__ Wih,      // [4*hid, in_dim] bf16
    const unsigned short* __restrict__ Whh,      // [4*hid, hid] bf16
    const float*          __restrict__ bias,     // [4*hid] (b_ih + b_hh)
    float*                __restrict__ cstate,   // [B, hid] f32, in-place
    unsigned short*       __restrict__ hout,     // [B, hid] bf16 (seq slice @ t)
    float*                __restrict__ hout_f32, // optional f32 output slice
    int in_dim, int hid)
{
  const int tid  = threadIdx.x;
  const int g    = tid >> 5;           // gate index = wave index (wave32)
  const int lane = tid & 31;
  const int half = lane >> 4;
  const int r    = lane & 15;
  const int j32  = blockIdx.x * 32;    // 32 hidden units per block
  const int m0   = blockIdx.y * 16;    // batch tile

  const int rowW0 = g * hid + j32 + r;      // weight rows for the two B-frags
  const int rowA  = m0 + r;                 // activation row for the A-frag

  const unsigned short* wih0 = Wih + (size_t)rowW0 * in_dim;
  const unsigned short* wih1 = wih0 + (size_t)16 * in_dim;
  const unsigned short* whh0 = Whh + (size_t)rowW0 * hid;
  const unsigned short* whh1 = whh0 + (size_t)16 * hid;
  const unsigned short* inA  = in_t  + (size_t)rowA * in_dim;
  const unsigned short* hpA  = hprev + (size_t)rowA * hid;

  v8f acc0 = {};
  v8f acc1 = {};

  // Phase 1: input contribution  x_t @ W_ih^T
  gemm2_pipelined(acc0, acc1, inA, wih0, wih1, in_dim, half);
  // Phase 2: recurrent contribution  h_{t-1} @ W_hh^T
  gemm2_pipelined(acc0, acc1, hpA, whh0, whh1, hid, half);

  const float bval0 = bias[rowW0];
  const float bval1 = bias[rowW0 + 16];

  // Exchange the gate tiles through LDS (C/D layout: m = v+8*half, n = r)
  __shared__ float gl[4][16][32];
#pragma unroll
  for (int v = 0; v < 8; ++v) {
    gl[g][v + 8 * half][r]      = acc0[v] + bval0;
    gl[g][v + 8 * half][16 + r] = acc1[v] + bval1;
  }
  __syncthreads();

  // Cell update: 16 rows x 32 cols = 512 elements over 128 threads
  for (int e = tid; e < 512; e += 128) {
    const int m = e >> 5, n = e & 31;
    const float gi = gl[0][m][n];
    const float gf = gl[1][m][n];
    const float gg = gl[2][m][n];
    const float go = gl[3][m][n];
    const float si = 1.0f / (1.0f + __expf(-gi));
    const float sf = 1.0f / (1.0f + __expf(-gf));
    const float so = 1.0f / (1.0f + __expf(-go));
    const size_t idx = (size_t)(m0 + m) * hid + (j32 + n);
    const float cn = sf * cstate[idx] + si * tanhf(gg);
    const float h  = so * tanhf(cn);
    cstate[idx] = cn;
    hout[idx]   = f32_to_bf16(h);
    if (hout_f32) hout_f32[idx] = h;
  }
}

// ---------------------------------------------------------------------------
// Utility kernels
// ---------------------------------------------------------------------------
__global__ void cast_f32_bf16_kernel(const float* __restrict__ src,
                                     unsigned short* __restrict__ dst, long n) {
  long i = (long)blockIdx.x * blockDim.x + threadIdx.x;
  const long stride = (long)gridDim.x * blockDim.x;
  for (; i < n; i += stride) dst[i] = f32_to_bf16(src[i]);
}

__global__ void add_bias_kernel(const float* __restrict__ a,
                                const float* __restrict__ b,
                                float* __restrict__ o, long n) {
  long i = (long)blockIdx.x * blockDim.x + threadIdx.x;
  const long stride = (long)gridDim.x * blockDim.x;
  for (; i < n; i += stride) o[i] = a[i] + b[i];
}

__global__ void zero_f32_kernel(float* __restrict__ p, long n) {
  long i = (long)blockIdx.x * blockDim.x + threadIdx.x;
  const long stride = (long)gridDim.x * blockDim.x;
  for (; i < n; i += stride) p[i] = 0.0f;
}

__global__ void zero_u16_kernel(unsigned short* __restrict__ p, long n) {
  long i = (long)blockIdx.x * blockDim.x + threadIdx.x;
  const long stride = (long)gridDim.x * blockDim.x;
  for (; i < n; i += stride) p[i] = 0;
}

// ---------------------------------------------------------------------------
// Launcher
// ---------------------------------------------------------------------------
extern "C" void kernel_launch(void* const* d_in, const int* in_sizes, int n_in,
                              void* d_out, int out_size, void* d_ws, size_t ws_size,
                              hipStream_t stream) {
  (void)in_sizes; (void)n_in; (void)out_size; (void)ws_size;

  constexpr int kT = 128, kB = 256, kD = 1280, kH = 256, kL = 5;

  const float* x     = (const float*)d_in[0];
  const float* eWih0 = (const float*)d_in[1];   // [1024, 1280]
  const float* eWih  = (const float*)d_in[2];   // [4, 1024, 256]
  const float* eWhh  = (const float*)d_in[3];   // [5, 1024, 256]
  const float* eBih  = (const float*)d_in[4];   // [5, 1024]
  const float* eBhh  = (const float*)d_in[5];
  const float* dWih0 = (const float*)d_in[6];   // [5120, 256]
  const float* dWih  = (const float*)d_in[7];   // [4, 5120, 1280]
  const float* dWhh  = (const float*)d_in[8];   // [5, 5120, 1280]
  const float* dBih  = (const float*)d_in[9];   // [5, 5120]
  const float* dBhh  = (const float*)d_in[10];

  float* out = (float*)d_out;   // [T,B,256] then [T,B,1280], fp32

  // ---- workspace layout -------------------------------------------------
  char* ws = (char*)d_ws;
  size_t off = 0;
  auto alloc = [&](size_t bytes) -> char* {
    char* p = ws + off;
    off = (off + bytes + 255) & ~(size_t)255;
    return p;
  };
  const size_t seqElems = (size_t)kT * kB * kD;            // max-width sequence
  unsigned short* seq0    = (unsigned short*)alloc(seqElems * 2);
  unsigned short* seq1    = (unsigned short*)alloc(seqElems * 2);
  unsigned short* x_bf    = (unsigned short*)alloc(seqElems * 2);
  unsigned short* eWih0_b = (unsigned short*)alloc((size_t)4 * kH * kD * 2);
  unsigned short* eWih_b  = (unsigned short*)alloc((size_t)(kL - 1) * 4 * kH * kH * 2);
  unsigned short* eWhh_b  = (unsigned short*)alloc((size_t)kL * 4 * kH * kH * 2);
  unsigned short* dWih0_b = (unsigned short*)alloc((size_t)4 * kD * kH * 2);
  unsigned short* dWih_b  = (unsigned short*)alloc((size_t)(kL - 1) * 4 * kD * kD * 2);
  unsigned short* dWhh_b  = (unsigned short*)alloc((size_t)kL * 4 * kD * kD * 2);
  float*          biasE   = (float*)alloc((size_t)kL * 4 * kH * 4);
  float*          biasD   = (float*)alloc((size_t)kL * 4 * kD * 4);
  float*          cbuf    = (float*)alloc((size_t)kB * kD * 4);
  unsigned short* hzero   = (unsigned short*)alloc((size_t)kB * kD * 2);

  // ---- one-time prep: bf16 casts, fused biases, zero h0 -----------------
  auto cast = [&](const float* s, unsigned short* d, long n) {
    cast_f32_bf16_kernel<<<4096, 256, 0, stream>>>(s, d, n);
  };
  cast(x,     x_bf,    (long)kT * kB * kD);
  cast(eWih0, eWih0_b, (long)4 * kH * kD);
  cast(eWih,  eWih_b,  (long)(kL - 1) * 4 * kH * kH);
  cast(eWhh,  eWhh_b,  (long)kL * 4 * kH * kH);
  cast(dWih0, dWih0_b, (long)4 * kD * kH);
  cast(dWih,  dWih_b,  (long)(kL - 1) * 4 * kD * kD);
  cast(dWhh,  dWhh_b,  (long)kL * 4 * kD * kD);
  add_bias_kernel<<<32, 256, 0, stream>>>(eBih, eBhh, biasE, (long)kL * 4 * kH);
  add_bias_kernel<<<128, 256, 0, stream>>>(dBih, dBhh, biasD, (long)kL * 4 * kD);
  zero_u16_kernel<<<1280, 256, 0, stream>>>(hzero, (long)kB * kD);

  unsigned short* bufs[2] = {seq0, seq1};
  int cur = 0;
  const unsigned short* seqIn = x_bf;   // encoder layer-0 input (bf16)

  // ---- encoder: 5 layers, hid = 256 ------------------------------------
  for (int l = 0; l < kL; ++l) {
    const int in_dim = (l == 0) ? kD : kH;
    const unsigned short* Wih =
        (l == 0) ? eWih0_b : eWih_b + (size_t)(l - 1) * 4 * kH * kH;
    const unsigned short* Whh = eWhh_b + (size_t)l * 4 * kH * kH;
    const float* bias = biasE + (size_t)l * 4 * kH;
    unsigned short* outSeq = bufs[cur];
    float* outF = (l == kL - 1) ? out : nullptr;

    zero_f32_kernel<<<256, 256, 0, stream>>>(cbuf, (long)kB * kH);
    const dim3 grid(kH / 32, kB / 16);
    for (int t = 0; t < kT; ++t) {
      const unsigned short* in_t = seqIn + (size_t)t * kB * in_dim;
      const unsigned short* hp =
          (t == 0) ? hzero : outSeq + (size_t)(t - 1) * kB * kH;
      unsigned short* ho = outSeq + (size_t)t * kB * kH;
      float* hoF = outF ? outF + (size_t)t * kB * kH : nullptr;
      lstm_step_kernel<<<grid, 128, 0, stream>>>(
          in_t, hp, Wih, Whh, bias, cbuf, ho, hoF, in_dim, kH);
    }
    seqIn = outSeq;
    cur ^= 1;
  }

  // ---- decoder: 5 layers, hid = 1280 ------------------------------------
  float* outDec = out + (size_t)kT * kB * kH;
  for (int l = 0; l < kL; ++l) {
    const int in_dim = (l == 0) ? kH : kD;
    const unsigned short* Wih =
        (l == 0) ? dWih0_b : dWih_b + (size_t)(l - 1) * 4 * kD * kD;
    const unsigned short* Whh = dWhh_b + (size_t)l * 4 * kD * kD;
    const float* bias = biasD + (size_t)l * 4 * kD;
    unsigned short* outSeq = bufs[cur];
    float* outF = (l == kL - 1) ? outDec : nullptr;

    zero_f32_kernel<<<1280, 256, 0, stream>>>(cbuf, (long)kB * kD);
    const dim3 grid(kD / 32, kB / 16);
    for (int t = 0; t < kT; ++t) {
      const unsigned short* in_t = seqIn + (size_t)t * kB * in_dim;
      const unsigned short* hp =
          (t == 0) ? hzero : outSeq + (size_t)(t - 1) * kB * kD;
      unsigned short* ho = outSeq + (size_t)t * kB * kD;
      float* hoF = outF ? outF + (size_t)t * kB * kD : nullptr;
      lstm_step_kernel<<<grid, 128, 0, stream>>>(
          in_t, hp, Wih, Whh, bias, cbuf, ho, hoF, in_dim, kD);
    }
    seqIn = outSeq;
    cur ^= 1;
  }
}
